// SRC_6416681140400
// MI455X (gfx1250) — compile-verified, matrix operand order checked by reading, not run on hardware
//
#include <hip/hip_runtime.h>

// ============================================================================
// MI455X (gfx1250) implementation.
//
// Dead-code analysis of the reference: blend2 (decoder LSTM) adds a per-batch
// constant to every candidate's logit -> softmax shift-invariance + argmax
// make the encoder LSTM, decoder LSTM, l1, targets and initial_logs all dead.
// The outputs reduce to:
//   X     = emb[origin_path] @ l2_W^T + l2_b                    [B*L, 512]
//   enc   = postLN_transformer(X) + X
//   w     = vt @ W1                                             [512]
//   score = enc . w                                             [B, L]
//   sels  = stable top-32 of score; paths = origin_path[sel];
//   probs[t] = 1 / sum_{unmasked} exp(score - score[sel_t])
// All GEMMs run on v_wmma_f32_16x16x32_bf16 (f32 accumulate), activations
// kept fp32 in HBM, converted to bf16 while staging into LDS.
// ============================================================================

typedef __attribute__((ext_vector_type(16))) __bf16 v16bf;
typedef __attribute__((ext_vector_type(8)))  float  v8f;

union Frag { v16bf v; unsigned int u[8]; };

__device__ __forceinline__ unsigned short f2bf(float f) {
  unsigned int u = __float_as_uint(f);
  u += 0x7FFFu + ((u >> 16) & 1u);   // round-to-nearest-even
  return (unsigned short)(u >> 16);
}
__device__ __forceinline__ unsigned int packbf(float a, float b) {
  return (unsigned int)f2bf(a) | ((unsigned int)f2bf(b) << 16);
}

// ---------------------------------------------------------------------------
// Generic WMMA GEMM:  C[M,N] = epilogue(A[M,K] @ B[K,N])
//   A_GATHER : A rows gathered from `gatherSrc` via Aidx (embedding lookup)
//   B_BF16   : B is pre-converted bf16, layout [N][K]
//   B_KN     : (f32 B only) layout [K][N] instead of [N][K]
//   BIAS/RELU/RES : epilogue terms
// Block tile 128x128, K-step 32, 8 waves of 32 lanes, wave tile 32x64.
// ---------------------------------------------------------------------------
template<bool A_GATHER, bool B_BF16, bool B_KN, bool BIAS, bool RELU, bool RES>
__global__ __launch_bounds__(256) void gemm_wmma(
    const float* __restrict__ A,
    const int*   __restrict__ Aidx,
    const float* __restrict__ gatherSrc,
    const void*  __restrict__ Bv,
    const float* __restrict__ bias,
    const float* __restrict__ Res,
    float* __restrict__ C,
    int M, int N, int K,
    long aBatch, long bBatch, long cBatch)
{
  constexpr int MBLK = 128, NBLK = 128, KT = 32, LDP = KT + 8; // pad: 80B rows
  __shared__ unsigned short AL[MBLK][LDP];   // [m][k] bf16
  __shared__ unsigned short BL[NBLK][LDP];   // [n][k] bf16

  const int tid  = threadIdx.x;
  const int z    = blockIdx.z;
  const int m0   = blockIdx.y * MBLK;
  const int n0   = blockIdx.x * NBLK;
  const int wid  = tid >> 5, lane = tid & 31;
  const int wm   = (wid >> 1) * 32;          // 0,32,64,96
  const int wn   = (wid & 1) * 64;           // 0,64
  const int lr   = lane & 15;
  const int ks   = (lane & 16) ? 8 : 0;      // ISA 16-bit A/B lane K offset
  const int mhi  = (lane & 16) ? 8 : 0;      // C/D lane M offset

  v8f acc[2][4] = {};

  // cooperative staging assignment: thread -> (row r, 16-elem half)
  const int r     = tid >> 1;                // 0..127
  const int cbase = (tid & 1) * 16;          // 0 or 16

  const float* arow;
  if (A_GATHER) arow = gatherSrc + (long)Aidx[m0 + r] * K;
  else          arow = A + aBatch * z + (long)(m0 + r) * K;

  for (int k0 = 0; k0 < K; k0 += KT) {
    // ---- stage A tile (f32 -> bf16) ----
    {
      const float* src = arow + k0 + cbase;
      if (k0 + KT < K) __builtin_prefetch(src + KT, 0, 0);
      const float4* s4 = (const float4*)src;
#pragma unroll
      for (int j = 0; j < 4; ++j) {
        float4 x = s4[j];
        *(unsigned int*)&AL[r][cbase + 4*j]     = packbf(x.x, x.y);
        *(unsigned int*)&AL[r][cbase + 4*j + 2] = packbf(x.z, x.w);
      }
    }
    // ---- stage B tile as [n][k] ----
    if (B_BF16) {
      const unsigned short* src =
          (const unsigned short*)Bv + (long)(n0 + r) * K + k0 + cbase;
      uint4 x0 = ((const uint4*)src)[0];
      uint4 x1 = ((const uint4*)src)[1];
      *(uint4*)&BL[r][cbase]     = x0;
      *(uint4*)&BL[r][cbase + 8] = x1;
    } else if (!B_KN) {
      const float* src =
          (const float*)Bv + bBatch * z + (long)(n0 + r) * K + k0 + cbase;
      const float4* s4 = (const float4*)src;
#pragma unroll
      for (int j = 0; j < 4; ++j) {
        float4 x = s4[j];
        *(unsigned int*)&BL[r][cbase + 4*j]     = packbf(x.x, x.y);
        *(unsigned int*)&BL[r][cbase + 4*j + 2] = packbf(x.z, x.w);
      }
    } else {
      // f32 [K][N] -> transpose into BL[n][k]
      const float* src =
          (const float*)Bv + bBatch * z + (long)(k0 + cbase) * N + (n0 + r);
#pragma unroll
      for (int j = 0; j < 16; ++j)
        BL[r][cbase + j] = f2bf(src[(long)j * N]);
    }
    __syncthreads();

    // ---- fragments per ISA 7.12.2 (16-bit A 16x32 layout) ----
    Frag af[2], bfr[4];
#pragma unroll
    for (int mt = 0; mt < 2; ++mt) {
      const unsigned short* row = &AL[wm + mt * 16 + lr][0];
#pragma unroll
      for (int i = 0; i < 4; ++i) {
        af[mt].u[i]     = *(const unsigned int*)(row + ks + 2 * i);
        af[mt].u[4 + i] = *(const unsigned int*)(row + 16 + ks + 2 * i);
      }
    }
#pragma unroll
    for (int nt = 0; nt < 4; ++nt) {
      const unsigned short* row = &BL[wn + nt * 16 + lr][0];
#pragma unroll
      for (int i = 0; i < 4; ++i) {
        bfr[nt].u[i]     = *(const unsigned int*)(row + ks + 2 * i);
        bfr[nt].u[4 + i] = *(const unsigned int*)(row + 16 + ks + 2 * i);
      }
    }
#pragma unroll
    for (int mt = 0; mt < 2; ++mt)
#pragma unroll
      for (int nt = 0; nt < 4; ++nt)
        acc[mt][nt] = __builtin_amdgcn_wmma_f32_16x16x32_bf16(
            false, af[mt].v, false, bfr[nt].v, (short)0, acc[mt][nt],
            false, false);
    __syncthreads();
  }

  // ---- epilogue ----
  const long coff = cBatch * z;
#pragma unroll
  for (int mt = 0; mt < 2; ++mt) {
#pragma unroll
    for (int nt = 0; nt < 4; ++nt) {
      const int col = n0 + wn + nt * 16 + lr;
      float bv = BIAS ? bias[col] : 0.0f;
#pragma unroll
      for (int i = 0; i < 8; ++i) {
        const int row = m0 + wm + mt * 16 + i + mhi;
        float v = acc[mt][nt][i] + bv;
        if (RES)  v += Res[(long)row * N + col];
        if (RELU) v = fmaxf(v, 0.0f);
        C[coff + (long)row * N + col] = v;
      }
    }
  }
}

// ---------------------------------------------------------------------------
__global__ void f32_to_bf16_kernel(const float* __restrict__ src,
                                   unsigned short* __restrict__ dst, int n) {
  int i = blockIdx.x * blockDim.x + threadIdx.x;
  if (i < n) dst[i] = f2bf(src[i]);
}

// w[d] = sum_j vt[j] * W1[j][d]   (vt @ W1, 512x512)
__global__ void compute_w_kernel(const float* __restrict__ W1,
                                 const float* __restrict__ vt,
                                 float* __restrict__ w) {
  int d = blockIdx.x * blockDim.x + threadIdx.x;
  if (d < 512) {
    float s = 0.f;
    for (int j = 0; j < 512; ++j) s += vt[j] * W1[j * 512 + d];
    w[d] = s;
  }
}

// softmax over rows of length 256 (one block per row), with pre-scale 1/sqrt(512)
__global__ void softmax_rows(float* __restrict__ S, float scale) {
  __shared__ float red[256];
  const int row = blockIdx.x, t = threadIdx.x;
  float* p = S + (long)row * 256;
  float v = p[t] * scale;
  red[t] = v; __syncthreads();
  for (int s = 128; s; s >>= 1) { if (t < s) red[t] = fmaxf(red[t], red[t + s]); __syncthreads(); }
  float m = red[0]; __syncthreads();
  float e = __expf(v - m);
  red[t] = e; __syncthreads();
  for (int s = 128; s; s >>= 1) { if (t < s) red[t] += red[t + s]; __syncthreads(); }
  p[t] = e / red[0];
}

// y = LN(x)*g + b, rows of 512, one 256-thread block per row
__global__ void layernorm_rows(const float* __restrict__ X,
                               const float* __restrict__ g,
                               const float* __restrict__ b,
                               float* __restrict__ Y) {
  __shared__ float red[256];
  const int row = blockIdx.x, t = threadIdx.x;
  const float* x = X + (long)row * 512;
  float a0 = x[t], a1 = x[t + 256];
  red[t] = a0 + a1; __syncthreads();
  for (int s = 128; s; s >>= 1) { if (t < s) red[t] += red[t + s]; __syncthreads(); }
  float mean = red[0] * (1.0f / 512.0f); __syncthreads();
  float d0 = a0 - mean, d1 = a1 - mean;
  red[t] = d0 * d0 + d1 * d1; __syncthreads();
  for (int s = 128; s; s >>= 1) { if (t < s) red[t] += red[t + s]; __syncthreads(); }
  float rs = rsqrtf(red[0] * (1.0f / 512.0f) + 1e-5f);
  float* y = Y + (long)row * 512;
  y[t]       = d0 * rs * g[t]       + b[t];
  y[t + 256] = d1 * rs * g[t + 256] + b[t + 256];
}

// score[row] = (LN(t2)*g + b + X) . w   (rows of 512)
__global__ void ln_score_rows(const float* __restrict__ T2,
                              const float* __restrict__ X,
                              const float* __restrict__ g,
                              const float* __restrict__ b,
                              const float* __restrict__ w,
                              float* __restrict__ score) {
  __shared__ float red[256];
  const int row = blockIdx.x, t = threadIdx.x;
  const float* x = T2 + (long)row * 512;
  float a0 = x[t], a1 = x[t + 256];
  red[t] = a0 + a1; __syncthreads();
  for (int s = 128; s; s >>= 1) { if (t < s) red[t] += red[t + s]; __syncthreads(); }
  float mean = red[0] * (1.0f / 512.0f); __syncthreads();
  float d0 = a0 - mean, d1 = a1 - mean;
  red[t] = d0 * d0 + d1 * d1; __syncthreads();
  for (int s = 128; s; s >>= 1) { if (t < s) red[t] += red[t + s]; __syncthreads(); }
  float rs = rsqrtf(red[0] * (1.0f / 512.0f) + 1e-5f); __syncthreads();
  const float* xr = X + (long)row * 512;
  float part = (d0 * rs * g[t]       + b[t]       + xr[t])       * w[t] +
               (d1 * rs * g[t + 256] + b[t + 256] + xr[t + 256]) * w[t + 256];
  red[t] = part; __syncthreads();
  for (int s = 128; s; s >>= 1) { if (t < s) red[t] += red[t + s]; __syncthreads(); }
  if (t == 0) score[row] = red[0];
}

// Per-batch greedy decode: stable argmax top-n + masked-softmax prob of pick.
__global__ void topk_decode(const float* __restrict__ score,
                            const int* __restrict__ origin_path,
                            const int* __restrict__ nptr,
                            float* __restrict__ out, int B) {
  __shared__ float sv[256];
  __shared__ int   msk[256];
  __shared__ float rv[256];
  __shared__ int   ri[256];
  const int b = blockIdx.x, t = threadIdx.x;
  const int nstep = nptr[0];
  sv[t] = score[(long)b * 256 + t];
  msk[t] = 0;
  __syncthreads();
  for (int step = 0; step < nstep; ++step) {
    float v = msk[t] ? -1.0e9f : sv[t];
    rv[t] = v; ri[t] = t; __syncthreads();
    for (int s = 128; s; s >>= 1) {
      if (t < s) {
        float o = rv[t + s]; int oi = ri[t + s];
        if (o > rv[t] || (o == rv[t] && oi < ri[t])) { rv[t] = o; ri[t] = oi; }
      }
      __syncthreads();
    }
    int   sel  = ri[0];
    float smax = rv[0];
    __syncthreads();
    float e = msk[t] ? 0.0f : __expf(sv[t] - smax);
    rv[t] = e; __syncthreads();
    for (int s = 128; s; s >>= 1) { if (t < s) rv[t] += rv[t + s]; __syncthreads(); }
    if (t == 0) {
      float prob = 1.0f / rv[0];
      long base = (long)b * nstep + step;
      out[base]                        = (float)origin_path[(long)b * 256 + sel];
      out[(long)B * nstep + base]      = prob;
      out[2L * (long)B * nstep + base] = (float)sel;
      msk[sel] = 1;
    }
    __syncthreads();
  }
}

// ---------------------------------------------------------------------------
extern "C" void kernel_launch(void* const* d_in, const int* in_sizes, int n_in,
                              void* d_out, int out_size, void* d_ws, size_t ws_size,
                              hipStream_t stream) {
  (void)in_sizes; (void)n_in; (void)out_size; (void)ws_size;
  const int   B = 256, L = 256, M = B * L;   // 65536 rows
  const int   H = 512, IN = 256;

  const int*   origin_path = (const int*)d_in[3];
  const int*   nptr        = (const int*)d_in[4];
  const float* emb   = (const float*)d_in[5];
  const float* l2_W  = (const float*)d_in[8];
  const float* l2_b  = (const float*)d_in[9];
  const float* W1    = (const float*)d_in[18];
  const float* vt    = (const float*)d_in[20];
  const float* tq_W  = (const float*)d_in[21];
  const float* tq_b  = (const float*)d_in[22];
  const float* tk_W  = (const float*)d_in[23];
  const float* tk_b  = (const float*)d_in[24];
  const float* tv_W  = (const float*)d_in[25];
  const float* tv_b  = (const float*)d_in[26];
  const float* to_W  = (const float*)d_in[27];
  const float* to_b  = (const float*)d_in[28];
  const float* ln1_g = (const float*)d_in[29];
  const float* ln1_b = (const float*)d_in[30];
  const float* ff1_W = (const float*)d_in[31];
  const float* ff1_b = (const float*)d_in[32];
  const float* ff2_W = (const float*)d_in[33];
  const float* ff2_b = (const float*)d_in[34];
  const float* ln2_g = (const float*)d_in[35];
  const float* ln2_b = (const float*)d_in[36];

  // workspace carve-up
  char* p = (char*)d_ws;
  auto carve = [&](size_t bytes) -> void* {
    void* q = (void*)p; p += (bytes + 255) & ~(size_t)255; return q;
  };
  const size_t ACT = (size_t)M * H * sizeof(float);          // 128 MiB
  unsigned short* Wl2b = (unsigned short*)carve((size_t)H * IN * 2);
  unsigned short* Wqb  = (unsigned short*)carve((size_t)H * H * 2);
  unsigned short* Wkb  = (unsigned short*)carve((size_t)H * H * 2);
  unsigned short* Wvb  = (unsigned short*)carve((size_t)H * H * 2);
  unsigned short* Wob  = (unsigned short*)carve((size_t)H * H * 2);
  unsigned short* Wf1b = (unsigned short*)carve((size_t)H * H * 2);
  unsigned short* Wf2b = (unsigned short*)carve((size_t)H * H * 2);
  float* wvec  = (float*)carve(H * sizeof(float));
  float* Xf    = (float*)carve(ACT);      // inputs (kept: residual + score)
  float* Buf1  = (float*)carve(ACT);      // Q -> attnO -> ff1
  float* Buf2  = (float*)carve(ACT);      // K -> t1 -> t2
  float* Buf3  = (float*)carve(ACT);      // V -> y1
  float* Sbuf  = (float*)carve((size_t)B * L * L * sizeof(float)); // 64 MiB
  float* score = (float*)carve((size_t)M * sizeof(float));

  // 1) weights -> bf16 ([N][K] layout preserved); w = vt @ W1
  f32_to_bf16_kernel<<<(H * IN + 255) / 256, 256, 0, stream>>>(l2_W, Wl2b, H * IN);
  f32_to_bf16_kernel<<<(H * H + 255) / 256, 256, 0, stream>>>(tq_W, Wqb, H * H);
  f32_to_bf16_kernel<<<(H * H + 255) / 256, 256, 0, stream>>>(tk_W, Wkb, H * H);
  f32_to_bf16_kernel<<<(H * H + 255) / 256, 256, 0, stream>>>(tv_W, Wvb, H * H);
  f32_to_bf16_kernel<<<(H * H + 255) / 256, 256, 0, stream>>>(to_W, Wob, H * H);
  f32_to_bf16_kernel<<<(H * H + 255) / 256, 256, 0, stream>>>(ff1_W, Wf1b, H * H);
  f32_to_bf16_kernel<<<(H * H + 255) / 256, 256, 0, stream>>>(ff2_W, Wf2b, H * H);
  compute_w_kernel<<<2, 256, 0, stream>>>(W1, vt, wvec);

  const dim3 blk(256);
  // 2) X = emb[origin_path] @ l2^T + l2_b
  gemm_wmma<true, true, false, true, false, false>
      <<<dim3(H / 128, M / 128, 1), blk, 0, stream>>>(
      nullptr, origin_path, emb, Wl2b, l2_b, nullptr, Xf, M, H, IN, 0, 0, 0);
  // 3) Q, K, V projections
  gemm_wmma<false, true, false, true, false, false>
      <<<dim3(H / 128, M / 128, 1), blk, 0, stream>>>(
      Xf, nullptr, nullptr, Wqb, tq_b, nullptr, Buf1, M, H, H, 0, 0, 0);
  gemm_wmma<false, true, false, true, false, false>
      <<<dim3(H / 128, M / 128, 1), blk, 0, stream>>>(
      Xf, nullptr, nullptr, Wkb, tk_b, nullptr, Buf2, M, H, H, 0, 0, 0);
  gemm_wmma<false, true, false, true, false, false>
      <<<dim3(H / 128, M / 128, 1), blk, 0, stream>>>(
      Xf, nullptr, nullptr, Wvb, tv_b, nullptr, Buf3, M, H, H, 0, 0, 0);
  // 4) S = Q @ K^T per batch  (B f32 in [N][K] layout = K rows)
  gemm_wmma<false, false, false, false, false, false>
      <<<dim3(L / 128, L / 128, B), blk, 0, stream>>>(
      Buf1, nullptr, nullptr, Buf2, nullptr, nullptr, Sbuf,
      L, L, H, (long)L * H, (long)L * H, (long)L * L);
  softmax_rows<<<B * L, 256, 0, stream>>>(Sbuf, 0.044194173824159216f); // 1/sqrt(512)
  // 5) O = P @ V per batch (B f32 in [K][N] layout) -> Buf1 (Q dead)
  gemm_wmma<false, false, true, false, false, false>
      <<<dim3(H / 128, L / 128, B), blk, 0, stream>>>(
      Sbuf, nullptr, nullptr, Buf3, nullptr, nullptr, Buf1,
      L, H, L, (long)L * L, (long)L * H, (long)L * H);
  // 6) t1 = O @ Wo^T + to_b + X -> Buf2 (K dead)
  gemm_wmma<false, true, false, true, false, true>
      <<<dim3(H / 128, M / 128, 1), blk, 0, stream>>>(
      Buf1, nullptr, nullptr, Wob, to_b, Xf, Buf2, M, H, H, 0, 0, 0);
  // 7) y1 = LN1(t1) -> Buf3 (V dead)
  layernorm_rows<<<M, 256, 0, stream>>>(Buf2, ln1_g, ln1_b, Buf3);
  // 8) f1 = relu(y1 @ Wf1^T + b) -> Buf1
  gemm_wmma<false, true, false, true, true, false>
      <<<dim3(H / 128, M / 128, 1), blk, 0, stream>>>(
      Buf3, nullptr, nullptr, Wf1b, ff1_b, nullptr, Buf1, M, H, H, 0, 0, 0);
  // 9) t2 = f1 @ Wf2^T + b + y1 -> Buf2
  gemm_wmma<false, true, false, true, false, true>
      <<<dim3(H / 128, M / 128, 1), blk, 0, stream>>>(
      Buf1, nullptr, nullptr, Wf2b, ff2_b, Buf3, Buf2, M, H, H, 0, 0, 0);
  // 10) score = (LN2(t2) + X) . w
  ln_score_rows<<<M, 256, 0, stream>>>(Buf2, Xf, ln2_g, ln2_b, wvec, score);
  // 11) greedy decode -> (paths, probs, sels) as float
  topk_decode<<<B, 256, 0, stream>>>(score, origin_path, nptr, (float*)d_out, B);
}
